// MANN_3315714752715
// MI455X (gfx1250) — compile-verified, matrix operand order checked by reading, not run on hardware
//
#include <hip/hip_runtime.h>
#include <hip/hip_bf16.h>

// ---- model dims ----
#define B_    256
#define T_    50
#define H_    256
#define MS_   256
#define MD_   64
#define NR_   4
#define NC_   5
#define KX    416          // padded [input(400)|offset(5)] width
#define KBIG  928          // KX + 256 (r) + 256 (h)
#define NBIG  1024         // 4*H
#define NKP   272          // padded 260 (NR*MD + NR)
#define GAMMA_ 0.95f
#define EPS_   1e-8f
#define CTRL_THREADS 512   // 16 waves

typedef __attribute__((ext_vector_type(16))) __bf16 v16bf;
typedef __attribute__((ext_vector_type(8)))  float  v8f;
typedef __hip_bfloat16 bf16;

__device__ __forceinline__ float sigmoidf_(float x) { return 1.f / (1.f + expf(-x)); }

// Load a 16x32 (rows x K) bf16 WMMA operand fragment from a row-major
// [16+][ld] bf16 matrix. Per CDNA5 ISA 16-bit operand layout:
// lane l: row = l&15, K-halves = (l>>4)*8 + {0..7} and 16 + (l>>4)*8 + {0..7}.
__device__ __forceinline__ v16bf load_frag(const bf16* base, int ld, int k0) {
  int lane = threadIdx.x & 31;
  int row  = lane & 15;
  int kh   = (lane >> 4) << 3;
  const bf16* p = base + row * ld + k0 + kh;
  union { uint4 u[2]; v16bf v; } f;
  f.u[0] = *reinterpret_cast<const uint4*>(p);
  f.u[1] = *reinterpret_cast<const uint4*>(p + 16);
  return f.v;
}

__device__ __forceinline__ v8f wmma_bf16_(v16bf a, v16bf b, v8f c) {
  return __builtin_amdgcn_wmma_f32_16x16x32_bf16(false, a, false, b, (short)0, c, false, false);
}

// ---------------- prep kernels ----------------
__global__ void k_fill(float* p, float v, int n) {
  int i = blockIdx.x * blockDim.x + threadIdx.x;
  if (i < n) p[i] = v;
}

__global__ void k_pack_x(const float* __restrict__ x, const int* __restrict__ tgt,
                         bf16* __restrict__ Xbf) {
  int i = blockIdx.x * blockDim.x + threadIdx.x;
  const int total = B_ * T_ * KX;
  if (i >= total) return;
  int k = i % KX; int bt = i / KX; int t = bt % T_; int b = bt / T_;
  float v = 0.f;
  if (k < 400) v = x[(b * T_ + t) * 400 + k];
  else if (k < 405) { int c = k - 400; if (t > 0 && tgt[b * T_ + t - 1] == c) v = 1.f; }
  Xbf[i] = __float2bfloat16(v);
}

// WbigT[n][k], n-major rows of K=928: k<416 -> W_ih[x rows 0..404] (pad 0),
// 416..671 -> W_ih[405..660] (r part), 672..927 -> W_hh (h part)
__global__ void k_pack_wbig(const float* __restrict__ W_ih, const float* __restrict__ W_hh,
                            bf16* __restrict__ W) {
  int i = blockIdx.x * blockDim.x + threadIdx.x;
  const int total = NBIG * KBIG;
  if (i >= total) return;
  int k = i % KBIG; int n = i / KBIG;
  float v = 0.f;
  if (k < KX)            { if (k < 405) v = W_ih[k * NBIG + n]; }
  else if (k < KX + H_)  v = W_ih[(405 + (k - KX)) * NBIG + n];
  else                   v = W_hh[(k - KX - H_) * NBIG + n];
  W[i] = __float2bfloat16(v);
}

__global__ void k_pack_wkp(const float* __restrict__ W_kp, bf16* __restrict__ W) {
  int i = blockIdx.x * blockDim.x + threadIdx.x;
  const int total = NKP * H_;
  if (i >= total) return;
  int k = i % H_; int n = i / H_;
  float v = (n < 260) ? W_kp[k * 260 + n] : 0.f;
  W[i] = __float2bfloat16(v);
}

// ---------------- per-step controller: GEMM + LSTM + key projection ----------------
// grid 16 (batch groups of 16), block 512 (16 waves). Dynamic LDS = 65536 + 8192 B.
// Each wave owns 4 N-tiles; K loop is double-buffered so B-tile loads for kt+1
// overlap the WMMA chain for kt.
__global__ void __launch_bounds__(CTRL_THREADS, 1)
k_step_controller(const bf16* __restrict__ Xbf, const bf16* __restrict__ WbigT,
                  const bf16* __restrict__ WkpT,
                  const float* __restrict__ b_lstm, const float* __restrict__ b_kp,
                  const float* __restrict__ rbuf, float* h, float* c,
                  float* kbuf, float* alphabuf, float* outs, int t) {
  extern __shared__ char smem[];
  bf16*  Abf  = (bf16*)smem;                   // phase 1: A [16][928] bf16
  float* gls  = (float*)smem;                  // phase 2: gates [16][1024] f32 (aliases A)
  float* pls  = (float*)smem;                  // phase 3: p [16][272] f32 (aliases gates)
  bf16*  habf = (bf16*)(smem + 16 * NBIG * 4); // [16][256] bf16 (separate region)

  int bg = blockIdx.x;
  int tid = threadIdx.x;
  int wave = tid >> 5;
  int lane = tid & 31, col = lane & 15, rbase = (lane >> 4) * 8;

  // build A = [x_t | r | h] as bf16 in LDS
  for (int i = tid; i < 16 * KBIG; i += CTRL_THREADS) {
    int m = i / KBIG, k = i % KBIG;
    int b = bg * 16 + m;
    if (k < KX) { Abf[i] = Xbf[((size_t)b * T_ + t) * KX + k]; continue; }
    float v = (k < KX + H_) ? rbuf[b * H_ + (k - KX)] : h[b * H_ + (k - KX - H_)];
    Abf[i] = __float2bfloat16(v);
  }
  __syncthreads();

  // gates GEMM: [16,928] @ [928,1024]; each of 16 waves owns 4 N-tiles
  v8f acc[4];
  v8f zero = {0.f, 0.f, 0.f, 0.f, 0.f, 0.f, 0.f, 0.f};
#pragma unroll
  for (int j = 0; j < 4; ++j) acc[j] = zero;

  const bf16* wb = WbigT + (size_t)(wave * 4) * 16 * KBIG;
  const size_t TS = (size_t)16 * KBIG;  // elements per 16-row N-tile block

  // double-buffered K loop over 29 k-tiles (tiles 0..27 in pairs, tile 28 in tail)
  v16bf a0  = load_frag(Abf, KBIG, 0);
  v16bf p00 = load_frag(wb + 0 * TS, KBIG, 0);
  v16bf p01 = load_frag(wb + 1 * TS, KBIG, 0);
  v16bf p02 = load_frag(wb + 2 * TS, KBIG, 0);
  v16bf p03 = load_frag(wb + 3 * TS, KBIG, 0);
#pragma unroll 1
  for (int kt = 0; kt < 28; kt += 2) {
    int k1 = (kt + 1) * 32;
    v16bf a1  = load_frag(Abf, KBIG, k1);
    v16bf p10 = load_frag(wb + 0 * TS, KBIG, k1);
    v16bf p11 = load_frag(wb + 1 * TS, KBIG, k1);
    v16bf p12 = load_frag(wb + 2 * TS, KBIG, k1);
    v16bf p13 = load_frag(wb + 3 * TS, KBIG, k1);
    acc[0] = wmma_bf16_(a0, p00, acc[0]);
    acc[1] = wmma_bf16_(a0, p01, acc[1]);
    acc[2] = wmma_bf16_(a0, p02, acc[2]);
    acc[3] = wmma_bf16_(a0, p03, acc[3]);
    int k2 = (kt + 2) * 32;
    a0  = load_frag(Abf, KBIG, k2);
    p00 = load_frag(wb + 0 * TS, KBIG, k2);
    p01 = load_frag(wb + 1 * TS, KBIG, k2);
    p02 = load_frag(wb + 2 * TS, KBIG, k2);
    p03 = load_frag(wb + 3 * TS, KBIG, k2);
    acc[0] = wmma_bf16_(a1, p10, acc[0]);
    acc[1] = wmma_bf16_(a1, p11, acc[1]);
    acc[2] = wmma_bf16_(a1, p12, acc[2]);
    acc[3] = wmma_bf16_(a1, p13, acc[3]);
  }
  acc[0] = wmma_bf16_(a0, p00, acc[0]);
  acc[1] = wmma_bf16_(a0, p01, acc[1]);
  acc[2] = wmma_bf16_(a0, p02, acc[2]);
  acc[3] = wmma_bf16_(a0, p03, acc[3]);

  __syncthreads();  // A dead; write gates (+bias) to LDS
#pragma unroll
  for (int j = 0; j < 4; ++j) {
    int n0 = (wave * 4 + j) * 16 + col;
    float bl = b_lstm[n0];
#pragma unroll
    for (int v = 0; v < 8; ++v) gls[(rbase + v) * NBIG + n0] = acc[j][v] + bl;
  }
  __syncthreads();

  // LSTM pointwise
  for (int i = tid; i < 16 * H_; i += CTRL_THREADS) {
    int m = i >> 8, hx = i & 255;
    int b = bg * 16 + m;
    float gi = gls[m * NBIG + hx];
    float gf = gls[m * NBIG + 256 + hx];
    float gg = gls[m * NBIG + 512 + hx];
    float go = gls[m * NBIG + 768 + hx];
    float cc = sigmoidf_(gf) * c[b * H_ + hx] + sigmoidf_(gi) * tanhf(gg);
    float hh = sigmoidf_(go) * tanhf(cc);
    c[b * H_ + hx] = cc;
    h[b * H_ + hx] = hh;
    outs[((size_t)t * B_ + b) * 512 + hx] = hh;
    habf[m * H_ + hx] = __float2bfloat16(hh);
  }
  __syncthreads();

  // key projection: [16,256] @ [256,272] ; 17 N-tiles over 16 waves (wave 0 takes tile 16)
  for (int jj = 0; jj < 2; ++jj) {
    int nt = wave + jj * 16;
    if (nt > 16) break;  // wave-uniform
    v8f pa = zero;
#pragma unroll 1
    for (int kt = 0; kt < H_ / 32; ++kt) {
      v16bf a = load_frag(habf, H_, kt * 32);
      v16bf bb = load_frag(WkpT + (size_t)nt * 16 * H_, H_, kt * 32);
      pa = wmma_bf16_(a, bb, pa);
    }
#pragma unroll
    for (int v = 0; v < 8; ++v) pls[(rbase + v) * NKP + nt * 16 + col] = pa[v];
  }
  __syncthreads();
  for (int i = tid; i < 16 * 260; i += CTRL_THREADS) {
    int m = i / 260, q = i % 260;
    int b = bg * 16 + m;
    float p = pls[m * NKP + q] + b_kp[q];
    if (q < 256) kbuf[b * 256 + q] = tanhf(p);
    else         alphabuf[b * NR_ + (q - 256)] = sigmoidf_(p);
  }
}

// ---------------- per-step memory update ----------------
__device__ __forceinline__ float block_reduce_max(float v, float* red) {
#pragma unroll
  for (int o = 16; o; o >>= 1) v = fmaxf(v, __shfl_xor(v, o, 32));
  if ((threadIdx.x & 31) == 0) red[threadIdx.x >> 5] = v;
  __syncthreads();
  float r = red[0];
#pragma unroll
  for (int i = 1; i < 8; ++i) r = fmaxf(r, red[i]);
  __syncthreads();
  return r;
}
__device__ __forceinline__ float block_reduce_sum(float v, float* red) {
#pragma unroll
  for (int o = 16; o; o >>= 1) v += __shfl_xor(v, o, 32);
  if ((threadIdx.x & 31) == 0) red[threadIdx.x >> 5] = v;
  __syncthreads();
  float r = red[0];
#pragma unroll
  for (int i = 1; i < 8; ++i) r += red[i];
  __syncthreads();
  return r;
}

// grid 256 (one WG per batch element), block 256. Dynamic LDS ~75 KB.
__global__ void k_step_memory(const float* __restrict__ kbuf, const float* __restrict__ alphabuf,
                              float* M, float* w_r, float* w_u,
                              float* rbuf, float* outs, int t) {
  extern __shared__ float sm[];
  float* Ml  = sm;                 // [256][64]
  float* wwl = Ml + MS_ * MD_;     // [4][256]
  float* wrl = wwl + NR_ * MS_;    // [4][256]
  float* wul = wrl + NR_ * MS_;    // [256]
  float* kl  = wul + MS_;          // [4][64]
  float* mn  = kl + NR_ * MD_;     // [256] inverse M-row norms
  float* kn  = mn + MS_;           // [4] inverse k norms
  float* red = kn + NR_;           // [8]
  int*   lui = (int*)(red + 8);    // [4]

  int b = blockIdx.x, tid = threadIdx.x;

  wul[tid] = w_u[b * MS_ + tid];
  kl[tid]  = kbuf[b * (NR_ * MD_) + tid];
  for (int j = 0; j < NR_; ++j) wrl[j * MS_ + tid] = w_r[(b * NR_ + j) * MS_ + tid];
  __syncthreads();

  // top-4 least-used (ties -> lowest index, matches stable top_k of -w_u)
  if (tid == 0) {
    unsigned used[8] = {0, 0, 0, 0, 0, 0, 0, 0};
    for (int rr = 0; rr < NR_; ++rr) {
      float best = 3.4e38f; int bi = 0;
      for (int n = 0; n < MS_; ++n) {
        if (used[n >> 5] & (1u << (n & 31))) continue;
        float v = wul[n];
        if (v < best) { best = v; bi = n; }
      }
      used[bi >> 5] |= 1u << (bi & 31);
      lui[rr] = bi;
    }
  }
  __syncthreads();
  int lu0 = lui[0];

  // write weights
  for (int j = 0; j < NR_; ++j) {
    float a = alphabuf[b * NR_ + j];
    wwl[j * MS_ + tid] = a * wrl[j * MS_ + tid] + (1.f - a) * ((tid == lui[j]) ? 1.f : 0.f);
  }
  __syncthreads();

  // erase least-used slot + additive write; compute row norms (thread = slot)
  {
    float er = (tid == lu0) ? 0.f : 1.f;
    float w0 = wwl[tid], w1 = wwl[MS_ + tid], w2 = wwl[2 * MS_ + tid], w3 = wwl[3 * MS_ + tid];
    float nrm = 0.f;
    float* Mg = M + ((size_t)b * MS_ + tid) * MD_;
#pragma unroll 4
    for (int d = 0; d < MD_; ++d) {
      float m = Mg[d] * er + w0 * kl[d] + w1 * kl[MD_ + d] + w2 * kl[2 * MD_ + d] + w3 * kl[3 * MD_ + d];
      Ml[tid * MD_ + d] = m;
      Mg[d] = m;
      nrm += m * m;
    }
    mn[tid] = 1.f / (sqrtf(nrm) + EPS_);
  }
  if (tid < NR_) {
    float s = 0.f;
    for (int d = 0; d < MD_; ++d) { float v = kl[tid * MD_ + d]; s += v * v; }
    kn[tid] = 1.f / (sqrtf(s) + EPS_);
  }
  __syncthreads();

  // cosine similarity (stored into wrl)
  {
    float s0 = 0, s1 = 0, s2 = 0, s3 = 0;
    for (int d = 0; d < MD_; ++d) {
      float m = Ml[tid * MD_ + d];
      s0 += m * kl[d]; s1 += m * kl[MD_ + d]; s2 += m * kl[2 * MD_ + d]; s3 += m * kl[3 * MD_ + d];
    }
    float mi = mn[tid];
    wrl[tid]           = s0 * mi * kn[0];
    wrl[MS_ + tid]     = s1 * mi * kn[1];
    wrl[2 * MS_ + tid] = s2 * mi * kn[2];
    wrl[3 * MS_ + tid] = s3 * mi * kn[3];
  }
  __syncthreads();

  // softmax over slots per read head
  for (int j = 0; j < NR_; ++j) {
    float v = wrl[j * MS_ + tid];
    float mx = block_reduce_max(v, red);
    float e = expf(v - mx);
    float s = block_reduce_sum(e, red);
    wrl[j * MS_ + tid] = e / s;
  }
  __syncthreads();

  for (int j = 0; j < NR_; ++j) w_r[(b * NR_ + j) * MS_ + tid] = wrl[j * MS_ + tid];

  // reads[j][d] = sum_n w_r[j][n] * M[n][d]   (thread: j = tid>>6, d = tid&63)
  {
    int j = tid >> 6, d = tid & 63;
    float acc = 0.f;
    for (int n = 0; n < MS_; ++n) acc += wrl[j * MS_ + n] * Ml[n * MD_ + d];
    rbuf[b * (NR_ * MD_) + tid] = acc;
    outs[((size_t)t * B_ + b) * 512 + 256 + tid] = acc;
  }
  // usage update
  {
    float wuv = GAMMA_ * wul[tid];
    for (int j = 0; j < NR_; ++j) wuv += wrl[j * MS_ + tid] + wwl[j * MS_ + tid];
    w_u[b * MS_ + tid] = wuv;
  }
}

// ---------------- output projection + softmax ----------------
__global__ void k_output(const float* __restrict__ outs, const float* __restrict__ W_out,
                         const float* __restrict__ b_out, float* __restrict__ probs) {
  int row = blockIdx.x * blockDim.x + threadIdx.x;
  if (row >= B_ * T_) return;
  int b = row / T_, t = row % T_;
  const float* o = outs + ((size_t)t * B_ + b) * 512;
  float l0 = b_out[0], l1 = b_out[1], l2 = b_out[2], l3 = b_out[3], l4 = b_out[4];
  for (int i = 0; i < 512; ++i) {
    float v = o[i];
    const float* w = W_out + i * NC_;
    l0 += v * w[0]; l1 += v * w[1]; l2 += v * w[2]; l3 += v * w[3]; l4 += v * w[4];
  }
  float mx = fmaxf(fmaxf(fmaxf(l0, l1), fmaxf(l2, l3)), l4);
  float e0 = expf(l0 - mx), e1 = expf(l1 - mx), e2 = expf(l2 - mx), e3 = expf(l3 - mx), e4 = expf(l4 - mx);
  float s = e0 + e1 + e2 + e3 + e4;
  float* pr = probs + (size_t)row * NC_;
  pr[0] = e0 / s; pr[1] = e1 / s; pr[2] = e2 / s; pr[3] = e3 / s; pr[4] = e4 / s;
}

extern "C" void kernel_launch(void* const* d_in, const int* in_sizes, int n_in,
                              void* d_out, int out_size, void* d_ws, size_t ws_size,
                              hipStream_t stream) {
  const float* input_var = (const float*)d_in[0];
  const int*   target    = (const int*)d_in[1];
  const float* W_ih      = (const float*)d_in[2];
  const float* W_hh      = (const float*)d_in[3];
  const float* b_lstm    = (const float*)d_in[4];
  const float* W_kp      = (const float*)d_in[5];
  const float* b_kp      = (const float*)d_in[6];
  const float* W_out     = (const float*)d_in[7];
  const float* b_out     = (const float*)d_in[8];
  float* probs = (float*)d_out;

  char* ws = (char*)d_ws;
  size_t off = 0;
  auto take = [&](size_t bytes) -> char* {
    char* p = ws + off;
    off += (bytes + 255) & ~(size_t)255;
    return p;
  };
  bf16*  Xbf   = (bf16*)take((size_t)B_ * T_ * KX * 2);
  bf16*  WbigT = (bf16*)take((size_t)NBIG * KBIG * 2);
  bf16*  WkpT  = (bf16*)take((size_t)NKP * H_ * 2);
  float* hbuf  = (float*)take((size_t)B_ * H_ * 4);
  float* cbuf  = (float*)take((size_t)B_ * H_ * 4);
  float* rbuf  = (float*)take((size_t)B_ * NR_ * MD_ * 4);
  float* wr    = (float*)take((size_t)B_ * NR_ * MS_ * 4);
  float* wu    = (float*)take((size_t)B_ * MS_ * 4);
  float* Mbuf  = (float*)take((size_t)B_ * MS_ * MD_ * 4);
  float* kbuf  = (float*)take((size_t)B_ * NR_ * MD_ * 4);
  float* albuf = (float*)take((size_t)B_ * NR_ * 4);
  float* outs  = (float*)take((size_t)T_ * B_ * 512 * 4);

  dim3 blk(256);
  auto g = [](int n) { return dim3((unsigned)((n + 255) / 256)); };

  k_pack_x<<<g(B_ * T_ * KX), blk, 0, stream>>>(input_var, target, Xbf);
  k_pack_wbig<<<g(NBIG * KBIG), blk, 0, stream>>>(W_ih, W_hh, WbigT);
  k_pack_wkp<<<g(NKP * H_), blk, 0, stream>>>(W_kp, WkpT);
  k_fill<<<g(B_ * H_), blk, 0, stream>>>(hbuf, 0.f, B_ * H_);
  k_fill<<<g(B_ * H_), blk, 0, stream>>>(cbuf, 0.f, B_ * H_);
  k_fill<<<g(B_ * NR_ * MD_), blk, 0, stream>>>(rbuf, 0.f, B_ * NR_ * MD_);
  k_fill<<<g(B_ * NR_ * MS_), blk, 0, stream>>>(wr, 1.f / MS_, B_ * NR_ * MS_);
  k_fill<<<g(B_ * MS_), blk, 0, stream>>>(wu, 1.f / MS_, B_ * MS_);
  k_fill<<<g(B_ * MS_ * MD_), blk, 0, stream>>>(Mbuf, 1e-6f, B_ * MS_ * MD_);

  const size_t smem_ctrl = 16 * NBIG * 4 + 16 * H_ * 2;  // 65536 + 8192
  const size_t smem_mem  = (MS_ * MD_ + 3 * NR_ * MS_ + 2 * MS_ + NR_ * MD_ + NR_ + 8) * 4 + NR_ * 4;

  for (int t = 0; t < T_; ++t) {
    k_step_controller<<<dim3(16), dim3(CTRL_THREADS), smem_ctrl, stream>>>(
        Xbf, WbigT, WkpT, b_lstm, b_kp, rbuf, hbuf, cbuf, kbuf, albuf, outs, t);
    k_step_memory<<<dim3(B_), blk, smem_mem, stream>>>(kbuf, albuf, Mbuf, wr, wu, rbuf, outs, t);
  }
  k_output<<<g(B_ * T_), blk, 0, stream>>>(outs, W_out, b_out, probs);

  (void)in_sizes; (void)n_in; (void)out_size; (void)ws_size;
}
